// StructureModule_65257733096121
// MI455X (gfx1250) — matrix operand order repaired and checked by reading, not compiled, not run on hardware
//
#include <hip/hip_runtime.h>
#include <hip/hip_bf16.h>

#define N_RES 1024
#define C_S 384
#define C_Z 128
#define C_H 16
#define H_HEADS 12
#define P_Q 4
#define P_V 8
#define HC (H_HEADS * C_H)            // 192
#define CAT_DIM 2112
#define INF_VAL 100000.0f
#define EPS_VAL 1e-8f

typedef __attribute__((ext_vector_type(16))) _Float16 v16h;
typedef __attribute__((ext_vector_type(8)))  float    v8f;
typedef __attribute__((ext_vector_type(4)))  int      v4i;

// ---------------------------------------------------------------------------
// Optional CDNA5 async global->LDS path (ASYNCcnt-tracked DMA, bypasses VGPRs)
// Builtin signature (from clang diagnostic): (as1 v4i*, as3 v4i*, imm, imm)
// ---------------------------------------------------------------------------
#if defined(__AMDGCN__) && __has_builtin(__builtin_amdgcn_global_load_async_to_lds_b128) && __has_builtin(__builtin_amdgcn_s_wait_asynccnt)
#define USE_ASYNC_LDS 1
typedef __attribute__((address_space(1))) v4i* as1_v4i_ptr;
typedef __attribute__((address_space(3))) v4i* as3_v4i_ptr;
#define ASYNC_B128(gsrc, ldst) \
    __builtin_amdgcn_global_load_async_to_lds_b128((as1_v4i_ptr)(gsrc), (as3_v4i_ptr)(ldst), 0, 0)
#define ASYNC_WAIT() __builtin_amdgcn_s_wait_asynccnt(0)
#else
#define USE_ASYNC_LDS 0
#endif

// ---------------------------------------------------------------------------
// WMMA fragment helpers (wave32, V_WMMA_F32_16X16X32_F16 layouts per ISA 7.12.2)
// ---------------------------------------------------------------------------

// A matrix 16x32 f16, LDS row-major [16][32] (row = M, col = K)
__device__ __forceinline__ v16h load_frag_A(const _Float16* lds, int lane) {
    const int row = lane & 15;
    const int grp = lane >> 4;
    v16h a;
#pragma unroll
    for (int e = 0; e < 16; ++e) {
        const int v = e >> 1, hh = e & 1;
        const int k = ((v >= 4) ? 16 : 0) + grp * 8 + (v & 3) * 2 + hh;
        a[e] = lds[row * 32 + k];
    }
    return a;
}

// A matrix from f32 LDS tile (convert at operand build)
__device__ __forceinline__ v16h load_frag_A_f32(const float* lds, int lane) {
    const int row = lane & 15;
    const int grp = lane >> 4;
    v16h a;
#pragma unroll
    for (int e = 0; e < 16; ++e) {
        const int v = e >> 1, hh = e & 1;
        const int k = ((v >= 4) ? 16 : 0) + grp * 8 + (v & 3) * 2 + hh;
        a[e] = (_Float16)lds[row * 32 + k];
    }
    return a;
}

// B matrix 32x16 f16, LDS row-major [32][ldb] (row = K, col = N)
__device__ __forceinline__ v16h load_frag_B(const _Float16* lds, int lane, int ldb) {
    const int col = lane & 15;
    const int kg  = lane >> 4;
    v16h b;
#pragma unroll
    for (int e = 0; e < 16; ++e) {
        const int v = e >> 1, hh = e & 1;
        const int k = ((v >= 4) ? 16 : 0) + kg * 8 + (v & 3) * 2 + hh;
        b[e] = lds[k * ldb + col];
    }
    return b;
}

// B matrix from f32 LDS tile
__device__ __forceinline__ v16h load_frag_B_f32(const float* lds, int lane, int ldb) {
    const int col = lane & 15;
    const int kg  = lane >> 4;
    v16h b;
#pragma unroll
    for (int e = 0; e < 16; ++e) {
        const int v = e >> 1, hh = e & 1;
        const int k = ((v >= 4) ? 16 : 0) + kg * 8 + (v & 3) * 2 + hh;
        b[e] = (_Float16)lds[k * ldb + col];
    }
    return b;
}

__device__ __forceinline__ float softplus_f(float x) { return log1pf(expf(x)); }

// ---------------------------------------------------------------------------
// Generic  out[M x Nout] = S[M x Kin] @ W[Nout x Kin]^T + bias   (WMMA f16)
// grid = (M/16, ceil(Nout/64)); block = 128 (4 waves, one 16-col tile each)
// ---------------------------------------------------------------------------
__global__ void gemm_xwt_bias(const float* __restrict__ S, const float* __restrict__ W,
                              const float* __restrict__ bias, float* __restrict__ out,
                              int Kin, int Nout) {
    __shared__ _Float16 As[16 * 32];
    __shared__ _Float16 Bs[4][32 * 16];
    const int lane = threadIdx.x & 31;
    const int wave = threadIdx.x >> 5;
    const int row0 = blockIdx.x * 16;
    const int col0 = (blockIdx.y * 4 + wave) * 16;
    const bool active = (col0 < Nout);
    v8f acc = {};
    for (int k0 = 0; k0 < Kin; k0 += 32) {
        for (int t = threadIdx.x; t < 512; t += 128) {
            const int r = t >> 5, k = t & 31;
            As[r * 32 + k] = (_Float16)S[(size_t)(row0 + r) * Kin + k0 + k];
        }
        if (active) {
            for (int t = lane; t < 512; t += 32) {
                const int k = t >> 4, n = t & 15;
                const int oc = col0 + n;
                float vv = (oc < Nout) ? W[(size_t)oc * Kin + k0 + k] : 0.f;
                Bs[wave][k * 16 + n] = (_Float16)vv;
            }
        }
        __syncthreads();
        if (active) {
            v16h a = load_frag_A(As, lane);
            v16h b = load_frag_B(Bs[wave], lane, 16);
            acc = __builtin_amdgcn_wmma_f32_16x16x32_f16(false, a, false, b,
                                                         (short)0, acc, false, false);
        }
        __syncthreads();
    }
    if (active) {
        const int col = lane & 15, grp = lane >> 4;
        const int oc = col0 + col;
        if (oc < Nout) {
            const float bv = bias ? bias[oc] : 0.f;
#pragma unroll
            for (int r = 0; r < 8; ++r)
                out[(size_t)(row0 + r + 8 * grp) * Nout + oc] = acc[r] + bv;
        }
    }
}

// ---------------------------------------------------------------------------
// Rigid frame: pts[n,t,:] = R[n] @ lin[n, (3,t)] + trans[n]
// ---------------------------------------------------------------------------
__global__ void make_points(const float* __restrict__ lin, const float* __restrict__ rot,
                            const float* __restrict__ trans, float* __restrict__ pts,
                            int npts) {
    const int idx = blockIdx.x * blockDim.x + threadIdx.x;
    if (idx >= N_RES * npts) return;
    const int n = idx / npts, t = idx - n * npts;
    const float lx = lin[(size_t)n * (3 * npts) + 0 * npts + t];
    const float ly = lin[(size_t)n * (3 * npts) + 1 * npts + t];
    const float lz = lin[(size_t)n * (3 * npts) + 2 * npts + t];
    const float* R = rot + (size_t)n * 9;
    const float* T = trans + (size_t)n * 3;
    pts[(size_t)idx * 3 + 0] = R[0] * lx + R[1] * ly + R[2] * lz + T[0];
    pts[(size_t)idx * 3 + 1] = R[3] * lx + R[4] * ly + R[5] * lz + T[1];
    pts[(size_t)idx * 3 + 2] = R[6] * lx + R[7] * ly + R[8] * lz + T[2];
}

// squared point norms per (n,h)
__global__ void pt_sq(const float* __restrict__ q_pts, const float* __restrict__ kv_pts,
                      float* __restrict__ qsq, float* __restrict__ ksq) {
    const int idx = blockIdx.x * blockDim.x + threadIdx.x;
    if (idx >= N_RES * H_HEADS) return;
    const int n = idx / H_HEADS, h = idx - n * H_HEADS;
    float sq = 0.f, sk = 0.f;
#pragma unroll
    for (int p = 0; p < P_Q; ++p)
#pragma unroll
        for (int c = 0; c < 3; ++c) {
            float a = q_pts[((size_t)n * 48 + h * P_Q + p) * 3 + c];
            float b = kv_pts[((size_t)(n * H_HEADS + h) * 12 + p) * 3 + c];
            sq += a * a; sk += b * b;
        }
    qsq[idx] = sq; ksq[idx] = sk;
}

// ---------------------------------------------------------------------------
// logits[h,i,j] = B_SCALE * (z[i,j,:] . w_b[h,:] + b_b[h])        (z pass 1)
// grid = (N, N/64); block = 128; A rows = j (f32 tile, async-staged), B = heads
// ---------------------------------------------------------------------------
__global__ void bias_logits(const float* __restrict__ z, const float* __restrict__ w_b,
                            const float* __restrict__ b_b, float* __restrict__ logits,
                            float B_SCALE) {
    __shared__ float    Zs[4][16 * 32];   // per-wave f32 A tile
    __shared__ _Float16 Ws[32 * 16];
    const int lane = threadIdx.x & 31;
    const int wave = threadIdx.x >> 5;
    const int i  = blockIdx.x;
    const int j0 = (blockIdx.y * 4 + wave) * 16;
    const size_t NN = (size_t)N_RES * N_RES;
    v8f acc = {};
    for (int k0 = 0; k0 < C_Z; k0 += 32) {
        for (int t = threadIdx.x; t < 512; t += 128) {
            const int k = t >> 4, hh = t & 15;
            Ws[k * 16 + hh] = (_Float16)((hh < H_HEADS) ? w_b[(size_t)hh * C_Z + k0 + k] : 0.f);
        }
#if USE_ASYNC_LDS
        {
            const int r = lane >> 1, half = lane & 1;      // each lane: 16 contiguous floats
            const float* gsrc = z + ((size_t)i * N_RES + j0 + r) * C_Z + k0 + half * 16;
            float* ldst = &Zs[wave][r * 32 + half * 16];
#pragma unroll
            for (int u = 0; u < 4; ++u)
                ASYNC_B128(gsrc + u * 4, ldst + u * 4);
            ASYNC_WAIT();
        }
#else
        for (int t = lane; t < 512; t += 32) {
            const int r = t >> 5, k = t & 31;
            Zs[wave][r * 32 + k] = z[((size_t)i * N_RES + j0 + r) * C_Z + k0 + k];
        }
#endif
        __syncthreads();
        v16h a = load_frag_A_f32(Zs[wave], lane);
        v16h b = load_frag_B(Ws, lane, 16);
        acc = __builtin_amdgcn_wmma_f32_16x16x32_f16(false, a, false, b,
                                                     (short)0, acc, false, false);
        __syncthreads();
    }
    const int hh = lane & 15, grp = lane >> 4;
    if (hh < H_HEADS) {
        const float bb = b_b[hh];
#pragma unroll
        for (int r = 0; r < 8; ++r) {
            const int j = j0 + r + 8 * grp;
            logits[(size_t)hh * NN + (size_t)i * N_RES + j] = B_SCALE * (acc[r] + bb);
        }
    }
}

// ---------------------------------------------------------------------------
// logits += QK_SCALE*q.k + hw*qpts.kpts  - 0.5*hw*(|q|^2+|k|^2) + mask
// grid = (N/16, N/64, H); block = 128; one K=32 WMMA (16 ch + 12 pt dims + pad)
// ---------------------------------------------------------------------------
__global__ void qk_logits(const float* __restrict__ q_buf, const float* __restrict__ kv_buf,
                          const float* __restrict__ q_pts, const float* __restrict__ kv_pts,
                          const float* __restrict__ qsq, const float* __restrict__ ksq,
                          const float* __restrict__ head_w, const float* __restrict__ mask,
                          float* __restrict__ logits, float QK_SCALE, float PT_SCALE) {
    __shared__ _Float16 As[16 * 32];
    __shared__ _Float16 Bs[4][32 * 16];
    const int lane = threadIdx.x & 31;
    const int wave = threadIdx.x >> 5;
    const int i0 = blockIdx.x * 16;
    const int j0 = (blockIdx.y * 4 + wave) * 16;
    const int h  = blockIdx.z;
    const size_t NN = (size_t)N_RES * N_RES;
    const float hw = softplus_f(head_w[h]) * PT_SCALE;

    for (int t = threadIdx.x; t < 512; t += 128) {
        const int r = t >> 5, k = t & 31;
        const int i = i0 + r;
        float vv = 0.f;
        if (k < 16)       vv = QK_SCALE * q_buf[(size_t)i * HC + h * C_H + k];
        else if (k < 28) { const int d = k - 16;
                           vv = hw * q_pts[((size_t)i * 48 + h * P_Q + d / 3) * 3 + (d % 3)]; }
        As[r * 32 + k] = (_Float16)vv;
    }
    for (int t = lane; t < 512; t += 32) {
        const int k = t >> 4, n = t & 15;
        const int j = j0 + n;
        float vv = 0.f;
        if (k < 16)       vv = kv_buf[(size_t)j * (2 * HC) + h * 32 + k];           // k-part
        else if (k < 28) { const int d = k - 16;
                           vv = kv_pts[((size_t)(j * H_HEADS + h) * 12 + d / 3) * 3 + (d % 3)]; }
        Bs[wave][k * 16 + n] = (_Float16)vv;
    }
    __syncthreads();
    v8f acc = {};
    {
        v16h a = load_frag_A(As, lane);
        v16h b = load_frag_B(Bs[wave], lane, 16);
        acc = __builtin_amdgcn_wmma_f32_16x16x32_f16(false, a, false, b,
                                                     (short)0, acc, false, false);
    }
    const int col = lane & 15, grp = lane >> 4;
    const int j = j0 + col;
    const float kterm = -0.5f * hw * ksq[(size_t)j * H_HEADS + h];
    const float mj = mask[j];
#pragma unroll
    for (int r = 0; r < 8; ++r) {
        const int i = i0 + r + 8 * grp;
        const float term = acc[r] + kterm - 0.5f * hw * qsq[(size_t)i * H_HEADS + h]
                         + INF_VAL * (mask[i] * mj - 1.0f);
        logits[(size_t)h * NN + (size_t)i * N_RES + j] += term;
    }
}

// ---------------------------------------------------------------------------
// softmax over j; one block per (h,i) row
// ---------------------------------------------------------------------------
__global__ void softmax_rows(float* __restrict__ logits) {
    float* p = logits + (size_t)blockIdx.x * N_RES;
    __shared__ float red[256];
    float m = -INFINITY;
    for (int j = threadIdx.x; j < N_RES; j += 256) m = fmaxf(m, p[j]);
    red[threadIdx.x] = m; __syncthreads();
    for (int s = 128; s > 0; s >>= 1) {
        if (threadIdx.x < s) red[threadIdx.x] = fmaxf(red[threadIdx.x], red[threadIdx.x + s]);
        __syncthreads();
    }
    m = red[0]; __syncthreads();
    float sum = 0.f;
    for (int j = threadIdx.x; j < N_RES; j += 256) { float e = __expf(p[j] - m); p[j] = e; sum += e; }
    red[threadIdx.x] = sum; __syncthreads();
    for (int s = 128; s > 0; s >>= 1) {
        if (threadIdx.x < s) red[threadIdx.x] += red[threadIdx.x + s];
        __syncthreads();
    }
    const float inv = 1.0f / red[0];
    for (int j = threadIdx.x; j < N_RES; j += 256) p[j] *= inv;
}

// ---------------------------------------------------------------------------
// o = a@v, o_pt = a@v_pts : grid = (N/16, H); block = 96 (3 waves: 16+16+8 cols)
// ---------------------------------------------------------------------------
__global__ void attn_ov(const float* __restrict__ a_mat, const float* __restrict__ kv_buf,
                        const float* __restrict__ kv_pts, float* __restrict__ cat,
                        float* __restrict__ opt) {
    __shared__ _Float16 As[16 * 32];
    __shared__ _Float16 Bs[3][32 * 16];
    const int lane = threadIdx.x & 31;
    const int wave = threadIdx.x >> 5;
    const int i0 = blockIdx.x * 16;
    const int h  = blockIdx.y;
    const size_t NN = (size_t)N_RES * N_RES;
    v8f acc = {};
    for (int k0 = 0; k0 < N_RES; k0 += 32) {
        for (int t = threadIdx.x; t < 512; t += 96) {
            const int r = t >> 5, k = t & 31;
            As[r * 32 + k] = (_Float16)a_mat[(size_t)h * NN + (size_t)(i0 + r) * N_RES + k0 + k];
        }
        for (int t = lane; t < 512; t += 32) {
            const int k = t >> 4, n = t & 15;
            const int j = k0 + k;
            float vv = 0.f;
            if (wave == 0) {
                vv = kv_buf[(size_t)j * (2 * HC) + h * 32 + C_H + n];              // v-part
            } else {
                const int d = (wave - 1) * 16 + n;                                  // 0..23 of v_pts
                if (d < 24)
                    vv = kv_pts[((size_t)(j * H_HEADS + h) * 12 + P_Q + d / 3) * 3 + (d % 3)];
            }
            Bs[wave][k * 16 + n] = (_Float16)vv;
        }
        __syncthreads();
        v16h a = load_frag_A(As, lane);
        v16h b = load_frag_B(Bs[wave], lane, 16);
        acc = __builtin_amdgcn_wmma_f32_16x16x32_f16(false, a, false, b,
                                                     (short)0, acc, false, false);
        __syncthreads();
    }
    const int col = lane & 15, grp = lane >> 4;
#pragma unroll
    for (int r = 0; r < 8; ++r) {
        const int i = i0 + r + 8 * grp;
        if (wave == 0) {
            cat[(size_t)i * CAT_DIM + h * C_H + col] = acc[r];
        } else {
            const int d = (wave - 1) * 16 + col;
            if (d < 24) opt[((size_t)i * H_HEADS + h) * 24 + d] = acc[r];
        }
    }
}

// ---------------------------------------------------------------------------
// o_pair[i,h,c] = sum_j a[h,i,j] z[i,j,c]    (z pass 2)
// grid = N; block = 256 (8 waves cover 128 z-channels); z async-staged f32
// ---------------------------------------------------------------------------
__global__ void attn_opair(const float* __restrict__ a_mat, const float* __restrict__ z,
                           float* __restrict__ cat) {
    __shared__ _Float16 As[16 * 32];
    __shared__ float    Zs[32 * 128];     // 16 KB f32 tile of z
    const int lane = threadIdx.x & 31;
    const int wave = threadIdx.x >> 5;    // col tile 0..7
    const int i = blockIdx.x;
    const size_t NN = (size_t)N_RES * N_RES;
    const float* zrow = z + (size_t)i * N_RES * C_Z;
    v8f acc = {};
    for (int k0 = 0; k0 < N_RES; k0 += 32) {
        for (int t = threadIdx.x; t < 512; t += 256) {
            const int hh = t >> 5, k = t & 31;
            As[hh * 32 + k] = (_Float16)((hh < H_HEADS)
                ? a_mat[(size_t)hh * NN + (size_t)i * N_RES + k0 + k] : 0.f);
        }
#if USE_ASYNC_LDS
        {
            // tile is one contiguous 16 KB block: 64 B per thread, 4 x b128 DMA
            const float* gsrc = zrow + (size_t)k0 * C_Z + threadIdx.x * 16;
            float* ldst = &Zs[threadIdx.x * 16];
#pragma unroll
            for (int u = 0; u < 4; ++u)
                ASYNC_B128(gsrc + u * 4, ldst + u * 4);
            ASYNC_WAIT();
        }
#else
        if (k0 + 32 < N_RES)
            __builtin_prefetch(zrow + (size_t)(k0 + 32) * C_Z + threadIdx.x * 16, 0, 1);
        for (int t = threadIdx.x; t < 4096; t += 256) {
            const int k = t >> 7, c = t & 127;
            Zs[k * 128 + c] = zrow[(size_t)(k0 + k) * C_Z + c];
        }
#endif
        __syncthreads();
        v16h a = load_frag_A(As, lane);
        v16h b = load_frag_B_f32(&Zs[wave * 16], lane, 128);
        acc = __builtin_amdgcn_wmma_f32_16x16x32_f16(false, a, false, b,
                                                     (short)0, acc, false, false);
        __syncthreads();
    }
    const int col = lane & 15, grp = lane >> 4;
#pragma unroll
    for (int r = 0; r < 8; ++r) {
        const int hh = r + 8 * grp;
        if (hh < H_HEADS)
            cat[(size_t)i * CAT_DIM + 576 + hh * C_Z + wave * 16 + col] = acc[r];
    }
}

// ---------------------------------------------------------------------------
// invert frames, norms, scatter into cat
// ---------------------------------------------------------------------------
__global__ void finalize_opt(const float* __restrict__ opt, const float* __restrict__ rot,
                             const float* __restrict__ trans, float* __restrict__ cat) {
    const int idx = blockIdx.x * blockDim.x + threadIdx.x;   // n*96 + h*8 + p
    if (idx >= N_RES * H_HEADS * P_V) return;
    const int n = idx / 96, hp = idx - n * 96;
    const int h = hp / P_V, p = hp - h * P_V;
    const float* R = rot + (size_t)n * 9;
    const float* T = trans + (size_t)n * 3;
    const float dx = opt[((size_t)n * H_HEADS + h) * 24 + p * 3 + 0] - T[0];
    const float dy = opt[((size_t)n * H_HEADS + h) * 24 + p * 3 + 1] - T[1];
    const float dz = opt[((size_t)n * H_HEADS + h) * 24 + p * 3 + 2] - T[2];
    const float lx = R[0] * dx + R[3] * dy + R[6] * dz;      // R^T
    const float ly = R[1] * dx + R[4] * dy + R[7] * dz;
    const float lz = R[2] * dx + R[5] * dy + R[8] * dz;
    float* c = cat + (size_t)n * CAT_DIM;
    c[192 + 0 * 96 + hp] = lx;
    c[192 + 1 * 96 + hp] = ly;
    c[192 + 2 * 96 + hp] = lz;
    c[192 + 3 * 96 + hp] = sqrtf(lx * lx + ly * ly + lz * lz + EPS_VAL);
}

// ---------------------------------------------------------------------------
extern "C" void kernel_launch(void* const* d_in, const int* in_sizes, int n_in,
                              void* d_out, int out_size, void* d_ws, size_t ws_size,
                              hipStream_t stream) {
    const float* s      = (const float*)d_in[0];
    const float* z      = (const float*)d_in[1];
    const float* rot    = (const float*)d_in[2];
    const float* trans  = (const float*)d_in[3];
    const float* mask   = (const float*)d_in[4];
    const float* w_q    = (const float*)d_in[5];
    const float* b_q    = (const float*)d_in[6];
    const float* w_kv   = (const float*)d_in[7];
    const float* b_kv   = (const float*)d_in[8];
    const float* w_qp   = (const float*)d_in[9];
    const float* b_qp   = (const float*)d_in[10];
    const float* w_kvp  = (const float*)d_in[11];
    const float* b_kvp  = (const float*)d_in[12];
    const float* w_b    = (const float*)d_in[13];
    const float* b_b    = (const float*)d_in[14];
    const float* head_w = (const float*)d_in[15];
    const float* w_out  = (const float*)d_in[16];
    const float* b_out  = (const float*)d_in[17];
    float* out = (float*)d_out;

    const float QK_SCALE = sqrtf(1.0f / (3.0f * C_H));
    const float B_SCALE  = sqrtf(1.0f / 3.0f);
    const float PT_SCALE = sqrtf(1.0f / (3.0f * (P_Q * 9.0f / 2.0f)));

    float* ws = (float*)d_ws;
    float* q_buf   = ws; ws += (size_t)N_RES * HC;        // 1024x192
    float* kv_buf  = ws; ws += (size_t)N_RES * 2 * HC;    // 1024x384
    float* qp_lin  = ws; ws += (size_t)N_RES * 144;
    float* kvp_lin = ws; ws += (size_t)N_RES * 432;
    float* q_pts   = ws; ws += (size_t)N_RES * 48 * 3;
    float* kv_pts  = ws; ws += (size_t)N_RES * 144 * 3;
    float* qsq     = ws; ws += (size_t)N_RES * H_HEADS;
    float* ksq     = ws; ws += (size_t)N_RES * H_HEADS;
    float* opt     = ws; ws += (size_t)N_RES * H_HEADS * 24;
    float* cat     = ws; ws += (size_t)N_RES * CAT_DIM;
    float* logits  = ws; ws += (size_t)H_HEADS * N_RES * N_RES;   // 48 MB

    dim3 blk128(128);
    // projections
    gemm_xwt_bias<<<dim3(64, 3), blk128, 0, stream>>>(s, w_q,   b_q,   q_buf,   C_S, HC);
    gemm_xwt_bias<<<dim3(64, 6), blk128, 0, stream>>>(s, w_kv,  b_kv,  kv_buf,  C_S, 2 * HC);
    gemm_xwt_bias<<<dim3(64, 3), blk128, 0, stream>>>(s, w_qp,  b_qp,  qp_lin,  C_S, 144);
    gemm_xwt_bias<<<dim3(64, 7), blk128, 0, stream>>>(s, w_kvp, b_kvp, kvp_lin, C_S, 432);
    // rigid frames
    make_points<<<(N_RES * 48 + 255) / 256, 256, 0, stream>>>(qp_lin, rot, trans, q_pts, 48);
    make_points<<<(N_RES * 144 + 255) / 256, 256, 0, stream>>>(kvp_lin, rot, trans, kv_pts, 144);
    pt_sq<<<(N_RES * H_HEADS + 255) / 256, 256, 0, stream>>>(q_pts, kv_pts, qsq, ksq);
    // attention logits (z pass 1) + qk/pt terms, softmax
    bias_logits<<<dim3(N_RES, 16), blk128, 0, stream>>>(z, w_b, b_b, logits, B_SCALE);
    qk_logits<<<dim3(64, 16, H_HEADS), blk128, 0, stream>>>(q_buf, kv_buf, q_pts, kv_pts,
                                                            qsq, ksq, head_w, mask, logits,
                                                            QK_SCALE, PT_SCALE);
    softmax_rows<<<H_HEADS * N_RES, 256, 0, stream>>>(logits);
    // outputs
    attn_ov<<<dim3(64, H_HEADS), 96, 0, stream>>>(logits, kv_buf, kv_pts, cat, opt);
    attn_opair<<<N_RES, 256, 0, stream>>>(logits, z, cat);       // z pass 2
    finalize_opt<<<(N_RES * 96 + 255) / 256, 256, 0, stream>>>(opt, rot, trans, cat);
    // final projection
    gemm_xwt_bias<<<dim3(64, 6), blk128, 0, stream>>>(cat, w_out, b_out, out, CAT_DIM, C_S);
}
